// MambaMixer_60078002536791
// MI455X (gfx1250) — matrix operand order repaired and checked by reading, not compiled, not run on hardware
//
#include <hip/hip_runtime.h>
#include <hip/hip_bf16.h>

// ---------------- problem constants ----------------
#define BB 2
#define LL 2048
#define HH 768
#define DI 1536
#define NN 16
#define RR 48
#define KK 4
#define MM (BB * LL)        // 4096 tokens
#define PK_DT 64            // dt GEMM K padded 48 -> 64

typedef __attribute__((ext_vector_type(16))) __bf16 v16bf;
typedef __attribute__((ext_vector_type(8)))  __bf16 v8bf;
typedef __attribute__((ext_vector_type(8)))  float  v8f;

// f32 -> bf16 round-to-nearest-even (used only in prep/epilogue, not GEMM loops)
__device__ __forceinline__ __bf16 f2bf(float f) {
  union { float f; unsigned u; } v; v.f = f;
  unsigned r = (v.u + 0x7FFFu + ((v.u >> 16) & 1u)) >> 16;
  union { unsigned short s; __bf16 b; } o; o.s = (unsigned short)r;
  return o.b;
}

__device__ __forceinline__ float fsigmoid(float x) {
  return __builtin_amdgcn_rcpf(1.f + __expf(-x));   // v_rcp_f32, no div chain
}
__device__ __forceinline__ float fsilu(float x) { return x * fsigmoid(x); }
__device__ __forceinline__ float fsoftplus(float x) {
  return (x > 20.f) ? x : __logf(1.f + __expf(x));
}

// Load 16 bf16 (two 16B vectors, at p0 and p1) into a WMMA operand register set.
__device__ __forceinline__ v16bf ld16bf(const __bf16* __restrict__ p0,
                                        const __bf16* __restrict__ p1) {
  v8bf x = *(const v8bf*)p0;
  v8bf y = *(const v8bf*)p1;
  return __builtin_shufflevector(x, y, 0, 1, 2, 3, 4, 5, 6, 7,
                                 8, 9, 10, 11, 12, 13, 14, 15);
}
// A operand (16x32 bf16, row-major [M,K]): lane lo = M, halves at K = 8*hi and 16+8*hi
__device__ __forceinline__ v16bf ldA(const __bf16* __restrict__ arow, int kk0, int hi) {
  return ld16bf(arow + kk0 + 8 * hi, arow + kk0 + 16 + 8 * hi);
}
// B operand (32x16 bf16, weights pre-transposed [N,K]): lane lo = N, halves at K = 8*hi and 16+8*hi
__device__ __forceinline__ v16bf ldB(const __bf16* __restrict__ brow, int kk0, int hi) {
  return ld16bf(brow + kk0 + 8 * hi, brow + kk0 + 16 + 8 * hi);
}

// =====================================================================
// prep: f32 -> bf16 row-major copy (activations)
// =====================================================================
__global__ __launch_bounds__(256) void k_prep_x(const float* __restrict__ src,
                                                __bf16* __restrict__ dst, int total) {
  int i = blockIdx.x * 256 + threadIdx.x;
  if (i < total) dst[i] = f2bf(src[i]);
}

// prep: transpose-convert weight W[K,N] f32 -> Wt[N,padK] bf16 (zero pad K..padK)
__global__ __launch_bounds__(256) void k_prep_wt(const float* __restrict__ src,
                                                 __bf16* __restrict__ dst,
                                                 int K, int N, int padK) {
  int i = blockIdx.x * 256 + threadIdx.x;
  int total = N * padK;
  if (i >= total) return;
  int n = i / padK, k = i - n * padK;
  dst[i] = f2bf(k < K ? src[(size_t)k * N + n] : 0.f);
}

// =====================================================================
// GEMM 1: proj = xb[4096,768] @ Wt_in^T  (Wt_in is [3072,768] bf16)
// dual 16x16 N-tiles per wave. 24576 dual-tiles: 256(M) x 96(N2) -> 6144 blocks
// epilogue: col<DI -> hpre[(b*DI+d)*LL+l] ; col>=DI -> gsil = silu(gate)
// =====================================================================
__global__ __launch_bounds__(128) void k_gemm_inproj(
    const __bf16* __restrict__ A, const __bf16* __restrict__ Bt,
    float* __restrict__ hpre, float* __restrict__ gsil) {
  const int wave = threadIdx.x >> 5, lane = threadIdx.x & 31;
  const int t = blockIdx.x * 4 + wave;
  const int tm = t & 255, tn2 = t >> 8;
  const int hi = lane >> 4, lo = lane & 15;
  const int mrow = tm * 16 + lo;
  const int n0 = tn2 * 32 + lo, n1 = n0 + 16;
  const __bf16* __restrict__ arow = A + (size_t)mrow * HH;
  const __bf16* __restrict__ br0 = Bt + (size_t)n0 * HH;
  const __bf16* __restrict__ br1 = Bt + (size_t)n1 * HH;

  v8f acc0 = {}, acc1 = {};
  for (int kk0 = 0; kk0 < HH; kk0 += 32) {
    v16bf a = ldA(arow, kk0, hi);
    v16bf b0 = ldB(br0, kk0, hi);
    v16bf b1 = ldB(br1, kk0, hi);
    acc0 = __builtin_amdgcn_wmma_f32_16x16x32_bf16(false, a, false, b0,
                                                   (short)0, acc0, false, false);
    acc1 = __builtin_amdgcn_wmma_f32_16x16x32_bf16(false, a, false, b1,
                                                   (short)0, acc1, false, false);
  }
#pragma unroll
  for (int r = 0; r < 8; ++r) {
    int row = tm * 16 + 8 * hi + r;
    int b_ = row >> 11, l_ = row & (LL - 1);
    if (n0 < DI) hpre[((size_t)b_ * DI + n0) * LL + l_] = acc0[r];
    else         gsil[(size_t)row * DI + (n0 - DI)] = fsilu(acc0[r]);
    if (n1 < DI) hpre[((size_t)b_ * DI + n1) * LL + l_] = acc1[r];
    else         gsil[(size_t)row * DI + (n1 - DI)] = fsilu(acc1[r]);
  }
}

// =====================================================================
// Depthwise causal conv (K=4, left pad 3) + bias + SiLU
// writes hc f32 [B*DI, LL] (for scan) and hct bf16 [M, DI] (token-major, GEMM A)
// =====================================================================
__global__ __launch_bounds__(256) void k_conv_silu(
    const float* __restrict__ hpre, const float* __restrict__ conv_w,
    const float* __restrict__ conv_b, float* __restrict__ hc,
    __bf16* __restrict__ hct) {
  const int c = blockIdx.x;            // c = b*DI + d
  const int b_ = c / DI, d = c - b_ * DI;
  const float w0 = conv_w[d * KK + 0], w1 = conv_w[d * KK + 1];
  const float w2 = conv_w[d * KK + 2], w3 = conv_w[d * KK + 3];
  const float bias = conv_b[d];
  const float* __restrict__ src = hpre + (size_t)c * LL;
  float* __restrict__ dst = hc + (size_t)c * LL;
  __bf16* __restrict__ dstt = hct + (size_t)b_ * LL * DI + d;
  for (int l = threadIdx.x; l < LL; l += 256) {
    float s = bias;
    if (l >= 3) s += w0 * src[l - 3];
    if (l >= 2) s += w1 * src[l - 2];
    if (l >= 1) s += w2 * src[l - 1];
    s += w3 * src[l];
    float v = fsilu(s);
    dst[l] = v;
    dstt[(size_t)l * DI] = f2bf(v);
  }
}

// =====================================================================
// GEMM 2: ssm_p[4096,80] = hct[4096,1536] @ Wt_xp^T  (Wt_xp [80,1536] bf16)
// single tile per wave, 1280 tiles -> 320 blocks
// epilogue: f32 ssmp; cols<64 also staged bf16 into ssmp_bf[4096,64] (dt_raw, pad 0)
// =====================================================================
__global__ __launch_bounds__(128) void k_gemm_xproj(
    const __bf16* __restrict__ A, const __bf16* __restrict__ Bt,
    float* __restrict__ ssmp, __bf16* __restrict__ ssmp_bf) {
  const int wave = threadIdx.x >> 5, lane = threadIdx.x & 31;
  const int t = blockIdx.x * 4 + wave;       // 256(M) x 5(N)
  const int tm = t & 255, tn = t >> 8;
  const int hi = lane >> 4, lo = lane & 15;
  const int mrow = tm * 16 + lo;
  const int ncol = tn * 16 + lo;
  const __bf16* __restrict__ arow = A + (size_t)mrow * DI;
  const __bf16* __restrict__ brow = Bt + (size_t)ncol * DI;

  v8f acc = {};
  for (int kk0 = 0; kk0 < DI; kk0 += 32) {
    v16bf a = ldA(arow, kk0, hi);
    v16bf b = ldB(brow, kk0, hi);
    acc = __builtin_amdgcn_wmma_f32_16x16x32_bf16(false, a, false, b,
                                                  (short)0, acc, false, false);
  }
#pragma unroll
  for (int r = 0; r < 8; ++r) {
    int row = tm * 16 + 8 * hi + r;
    ssmp[(size_t)row * (RR + 2 * NN) + ncol] = acc[r];
    if (ncol < PK_DT)
      ssmp_bf[(size_t)row * PK_DT + ncol] = f2bf(ncol < RR ? acc[r] : 0.f);
  }
}

// =====================================================================
// GEMM 3: dt = softplus(ssmp_bf[4096,64] @ Wt_dt^T + b)  (Wt_dt [1536,64] bf16)
// dual N-tiles: 256(M) x 48(N2) -> 3072 blocks; store transposed dtT[(b*DI+d)*LL+l]
// =====================================================================
__global__ __launch_bounds__(128) void k_gemm_dt(
    const __bf16* __restrict__ A, const __bf16* __restrict__ Bt,
    const float* __restrict__ bias, float* __restrict__ dtT) {
  const int wave = threadIdx.x >> 5, lane = threadIdx.x & 31;
  const int t = blockIdx.x * 4 + wave;
  const int tm = t & 255, tn2 = t >> 8;
  const int hi = lane >> 4, lo = lane & 15;
  const int mrow = tm * 16 + lo;
  const int n0 = tn2 * 32 + lo, n1 = n0 + 16;
  const __bf16* __restrict__ arow = A + (size_t)mrow * PK_DT;
  const __bf16* __restrict__ br0 = Bt + (size_t)n0 * PK_DT;
  const __bf16* __restrict__ br1 = Bt + (size_t)n1 * PK_DT;

  v8f acc0 = {}, acc1 = {};
#pragma unroll
  for (int kk0 = 0; kk0 < PK_DT; kk0 += 32) {
    v16bf a = ldA(arow, kk0, hi);
    v16bf b0 = ldB(br0, kk0, hi);
    v16bf b1 = ldB(br1, kk0, hi);
    acc0 = __builtin_amdgcn_wmma_f32_16x16x32_bf16(false, a, false, b0,
                                                   (short)0, acc0, false, false);
    acc1 = __builtin_amdgcn_wmma_f32_16x16x32_bf16(false, a, false, b1,
                                                   (short)0, acc1, false, false);
  }
  const float bc0 = bias[n0], bc1 = bias[n1];
#pragma unroll
  for (int r = 0; r < 8; ++r) {
    int row = tm * 16 + 8 * hi + r;
    int b_ = row >> 11, l_ = row & (LL - 1);
    dtT[((size_t)b_ * DI + n0) * LL + l_] = fsoftplus(acc0[r] + bc0);
    dtT[((size_t)b_ * DI + n1) * LL + l_] = fsoftplus(acc1[r] + bc1);
  }
}

// =====================================================================
// Selective scan: 16-lane half-wave per (b,d) channel, lane = state n.
// state = exp(A*dt)*state + B*hc ; y = sum_n state*C ; fuse skip + gate.
// Writes y directly as bf16 for the output GEMM. 384 blocks x 128.
// =====================================================================
__global__ __launch_bounds__(128) void k_scan(
    const float* __restrict__ dtT, const float* __restrict__ hc,
    const float* __restrict__ ssmp, const float* __restrict__ gsil,
    const float* __restrict__ A_log, const float* __restrict__ D_param,
    __bf16* __restrict__ yb) {
  const int tid = threadIdx.x;
  const int c = blockIdx.x * 8 + (tid >> 4);     // channel = b*DI + d
  const int n = tid & 15;
  const int b_ = c / DI, d_ = c - b_ * DI;

  const float an = -__expf(A_log[d_ * NN + n]);
  const float Dd = D_param[d_];
  const float* __restrict__ dt_c = dtT + (size_t)c * LL;
  const float* __restrict__ hc_c = hc + (size_t)c * LL;
  const float* __restrict__ sp_b = ssmp + (size_t)b_ * LL * (RR + 2 * NN) + RR + n;
  const float* __restrict__ gs_b = gsil + (size_t)b_ * LL * DI + d_;
  __bf16* __restrict__ y_b = yb + (size_t)b_ * LL * DI + d_;

  float state = 0.f;
  for (int l0 = 0; l0 < LL; l0 += 4) {
    __builtin_prefetch(dt_c + l0 + 32, 0, 1);    // global_prefetch_b8
    __builtin_prefetch(hc_c + l0 + 32, 0, 1);
    float dt4[4], hc4[4];
    *(float4*)dt4 = *(const float4*)(dt_c + l0);
    *(float4*)hc4 = *(const float4*)(hc_c + l0);
#pragma unroll
    for (int j = 0; j < 4; ++j) {
      const int l = l0 + j;
      const float Bn = sp_b[(size_t)l * (RR + 2 * NN)];
      const float Cn = sp_b[(size_t)l * (RR + 2 * NN) + NN];
      state = __expf(an * dt4[j]) * state + Bn * hc4[j];
      float p = state * Cn;
      p += __shfl_xor(p, 1, 32);
      p += __shfl_xor(p, 2, 32);
      p += __shfl_xor(p, 4, 32);
      p += __shfl_xor(p, 8, 32);
      if (n == 0) {
        float y = (p + hc4[j] * Dd) * gs_b[(size_t)l * DI];
        y_b[(size_t)l * DI] = f2bf(y);
      }
    }
  }
}

// =====================================================================
// GEMM 4: out[4096,768] = yb[4096,1536] @ Wt_out^T  (Wt_out [768,1536] bf16)
// dual N-tiles: 256(M) x 24(N2) -> 1536 blocks
// =====================================================================
__global__ __launch_bounds__(128) void k_gemm_out(
    const __bf16* __restrict__ A, const __bf16* __restrict__ Bt,
    float* __restrict__ out) {
  const int wave = threadIdx.x >> 5, lane = threadIdx.x & 31;
  const int t = blockIdx.x * 4 + wave;
  const int tm = t & 255, tn2 = t >> 8;
  const int hi = lane >> 4, lo = lane & 15;
  const int mrow = tm * 16 + lo;
  const int n0 = tn2 * 32 + lo, n1 = n0 + 16;
  const __bf16* __restrict__ arow = A + (size_t)mrow * DI;
  const __bf16* __restrict__ br0 = Bt + (size_t)n0 * DI;
  const __bf16* __restrict__ br1 = Bt + (size_t)n1 * DI;

  v8f acc0 = {}, acc1 = {};
  for (int kk0 = 0; kk0 < DI; kk0 += 32) {
    v16bf a = ldA(arow, kk0, hi);
    v16bf b0 = ldB(br0, kk0, hi);
    v16bf b1 = ldB(br1, kk0, hi);
    acc0 = __builtin_amdgcn_wmma_f32_16x16x32_bf16(false, a, false, b0,
                                                   (short)0, acc0, false, false);
    acc1 = __builtin_amdgcn_wmma_f32_16x16x32_bf16(false, a, false, b1,
                                                   (short)0, acc1, false, false);
  }
#pragma unroll
  for (int r = 0; r < 8; ++r) {
    int row = tm * 16 + 8 * hi + r;
    out[(size_t)row * HH + n0] = acc0[r];
    out[(size_t)row * HH + n1] = acc1[r];
  }
}

// =====================================================================
extern "C" void kernel_launch(void* const* d_in, const int* in_sizes, int n_in,
                              void* d_out, int out_size, void* d_ws, size_t ws_size,
                              hipStream_t stream) {
  const float* x         = (const float*)d_in[0];
  const float* in_proj_w = (const float*)d_in[1];
  const float* conv_w    = (const float*)d_in[2];
  const float* conv_b    = (const float*)d_in[3];
  const float* x_proj_w  = (const float*)d_in[4];
  const float* dt_proj_w = (const float*)d_in[5];
  const float* dt_proj_b = (const float*)d_in[6];
  const float* out_proj_w= (const float*)d_in[7];
  const float* A_log     = (const float*)d_in[8];
  const float* D_param   = (const float*)d_in[9];
  float* out = (float*)d_out;

  const size_t TOK = (size_t)MM * DI;   // 6291456
  // f32 region
  float* fws  = (float*)d_ws;
  float* dtT  = fws;                    // [B*DI, LL]  (aliases hpre: hpre dead after conv)
  float* hpre = dtT;                    // [B*DI, LL]
  float* hc   = fws + TOK;              // [B*DI, LL]
  float* gsil = fws + 2 * TOK;          // [M, DI]
  float* ssmp = fws + 3 * TOK;          // [M, 80]
  // bf16 region
  __bf16* bws    = (__bf16*)(fws + 3 * TOK + (size_t)MM * (RR + 2 * NN));
  __bf16* xb     = bws;                               // [M, HH]
  __bf16* Wt_in  = xb + (size_t)MM * HH;              // [2DI, HH]
  __bf16* Wt_xp  = Wt_in + (size_t)(2 * DI) * HH;     // [80, DI]
  __bf16* Wt_dt  = Wt_xp + (size_t)(RR + 2 * NN) * DI;// [DI, 64]
  __bf16* Wt_out = Wt_dt + (size_t)DI * PK_DT;        // [HH, DI]
  __bf16* hct    = Wt_out + (size_t)HH * DI;          // [M, DI]
  __bf16* ssm_bf = hct + TOK;                         // [M, 64]
  __bf16* yb     = ssm_bf + (size_t)MM * PK_DT;       // [M, DI]

  // ---- prep: bf16 staging (conversion happens once, not per WMMA) ----
  k_prep_x<<<(MM * HH + 255) / 256, 256, 0, stream>>>(x, xb, MM * HH);
  k_prep_wt<<<((2 * DI) * HH + 255) / 256, 256, 0, stream>>>(in_proj_w, Wt_in, HH, 2 * DI, HH);
  k_prep_wt<<<((RR + 2 * NN) * DI + 255) / 256, 256, 0, stream>>>(x_proj_w, Wt_xp, DI, RR + 2 * NN, DI);
  k_prep_wt<<<(DI * PK_DT + 255) / 256, 256, 0, stream>>>(dt_proj_w, Wt_dt, RR, DI, PK_DT);
  k_prep_wt<<<(HH * DI + 255) / 256, 256, 0, stream>>>(out_proj_w, Wt_out, DI, HH, DI);

  // ---- pipeline ----
  k_gemm_inproj<<<6144, 128, 0, stream>>>(xb, Wt_in, hpre, gsil);
  k_conv_silu<<<BB * DI, 256, 0, stream>>>(hpre, conv_w, conv_b, hc, hct);
  k_gemm_xproj<<<320, 128, 0, stream>>>(hct, Wt_xp, ssmp, ssm_bf);
  k_gemm_dt<<<3072, 128, 0, stream>>>(ssm_bf, Wt_dt, dt_proj_b, dtT);
  k_scan<<<384, 128, 0, stream>>>(dtT, hc, ssmp, gsil, A_log, D_param, yb);
  k_gemm_out<<<1536, 128, 0, stream>>>(yb, Wt_out, out);
}
